// JKNet_55293408969099
// MI455X (gfx1250) — matrix operand order repaired and checked by reading, not compiled, not run on hardware
//
#include <hip/hip_runtime.h>
#include <hip/hip_bf16.h>

// ---------------- problem constants ----------------
#define Bsz    8192
#define Vdim   1000
#define Vp     1024
#define Kche   5
#define CF     5          // CL1_F
#define Ppad   640        // 625 padded (= 128*5)
#define Preal  625
#define HID    256
#define OUTC   10
#define NL     4
#define BN_EPS 1e-5f
#define Bc     1024       // batch chunk for Chebyshev stage
#define NCHUNK (Bsz / Bc)

typedef __attribute__((ext_vector_type(8)))  float  v8f;
typedef __attribute__((ext_vector_type(16))) __bf16 v16bf;

union FragA { v16bf v; uint4 q[2]; };
union FragB { v16bf v; unsigned int u[8]; };

__device__ __forceinline__ unsigned short f2bf(float f) {
  unsigned int u = __float_as_uint(f);
  unsigned int r = u + 0x7FFFu + ((u >> 16) & 1u);   // round-to-nearest-even
  return (unsigned short)(r >> 16);
}

// ---------------- pack kernels ----------------
__global__ void pack_Lr(const float* __restrict__ L, const float* __restrict__ lmax,
                        unsigned short* __restrict__ Lrbf) {
  int idx = blockIdx.x * 256 + threadIdx.x;
  int i = idx / Vp, j = idx % Vp;
  float v = 0.f;
  if (i < Vdim && j < Vdim)
    v = 2.f * L[(size_t)i * Vdim + j] / lmax[0] - (i == j ? 1.f : 0.f);
  Lrbf[idx] = f2bf(v);
}

__global__ void pack_fc1W(const float* __restrict__ W, unsigned short* __restrict__ Wb) {
  int idx = blockIdx.x * 256 + threadIdx.x;      // Ppad*HID
  int p = idx / HID, n = idx % HID;
  float v = (p < Preal) ? W[(size_t)n * Preal + p] : 0.f;
  Wb[idx] = f2bf(v);
}

__global__ void pack_lin1W(const float* __restrict__ W, unsigned short* __restrict__ Wb) {
  int idx = blockIdx.x * 256 + threadIdx.x;      // HID*HID
  int k = idx / HID, n = idx % HID;
  Wb[idx] = f2bf(W[(size_t)n * HID + k]);
}

// ---------------- x chunk transpose: x[B,V] -> x0[Vp,Bc] (f32 + bf16) ----------------
__global__ void transpose_pack(const float* __restrict__ x,
                               float* __restrict__ x0,
                               unsigned short* __restrict__ x0bf,
                               int chunk) {
  __shared__ float tile[32][33];
  int vb = blockIdx.x * 32, bb = blockIdx.y * 32;
  int tx = threadIdx.x, ty = threadIdx.y;        // block (32,8)
  #pragma unroll
  for (int i = 0; i < 4; i++) {
    int b = bb + ty + i * 8;
    int v = vb + tx;
    float val = (v < Vdim) ? x[(size_t)(chunk * Bc + b) * Vdim + v] : 0.f;
    tile[ty + i * 8][tx] = val;
  }
  __syncthreads();
  #pragma unroll
  for (int i = 0; i < 4; i++) {
    int v = vb + ty + i * 8;
    int b = bb + tx;
    float val = tile[tx][ty + i * 8];
    x0[(size_t)v * Bc + b]   = val;
    x0bf[(size_t)v * Bc + b] = f2bf(val);
  }
}

// ---------------- bf16 WMMA GEMM: C[M,N] = A[M,K] * B[K,N] ----------------
// Block tile 128(M) x 256(N), 8 waves in 2x4, wave tile 64x64 = 16 WMMAs/stage.
// flags: 1 = Chebyshev epilogue (out = 2*acc - prev2), 2 = +bias, 4 = relu
#define F_CHEB 1
#define F_BIAS 2
#define F_RELU 4

__global__ __launch_bounds__(256)
void gemm_bf16(const unsigned short* __restrict__ A, int lda,
               const unsigned short* __restrict__ Bm, int ldb,
               float* __restrict__ Cout, unsigned short* __restrict__ Cbf, int ldc,
               const float* __restrict__ prev2, const float* __restrict__ bias,
               int flags, int Kd) {
  __shared__ unsigned short As[128 * 40];   // 128 rows x 32 halves, stride 40 (16B aligned)
  __shared__ unsigned int   Bs[16 * 256];   // 16 K-pair dword rows x 256 columns

  const int tid   = threadIdx.x;
  const int wid   = tid >> 5, lane = tid & 31;
  const int wm    = (wid & 1) * 64;         // wave M offset in block tile
  const int wn    = (wid >> 1) * 64;        // wave N offset in block tile
  const int lhalf = lane >> 4;              // 0: lanes 0-15, 1: lanes 16-31
  const int lidx  = lane & 15;
  const int mBlk  = blockIdx.y * 128, nBlk = blockIdx.x * 256;

  const int ar = tid >> 1, aseg = tid & 1;        // A stage: row, 16-half segment
  const int bp = tid >> 4, bn = (tid & 15) * 16;  // B stage: k-pair, 16-col segment

  v8f acc[4][4];
  #pragma unroll
  for (int i = 0; i < 4; i++)
    #pragma unroll
    for (int j = 0; j < 4; j++) acc[i][j] = (v8f)0.f;

  for (int kb = 0; kb < Kd; kb += 32) {
    // ---- global fetch ----
    const uint4* ag = (const uint4*)(A + (size_t)(mBlk + ar) * lda + kb + aseg * 16);
    uint4 a0 = ag[0], a1 = ag[1];
    const uint4* bg0 = (const uint4*)(Bm + (size_t)(kb + 2 * bp) * ldb + nBlk + bn);
    const uint4* bg1 = (const uint4*)(Bm + (size_t)(kb + 2 * bp + 1) * ldb + nBlk + bn);
    uint4 b00 = bg0[0], b01 = bg0[1];      // row k,   16 halves
    uint4 b10 = bg1[0], b11 = bg1[1];      // row k+1, 16 halves
    if (kb + 32 < Kd)  // prefetch next A tile (global_prefetch_b8)
      __builtin_prefetch((const char*)(A + (size_t)(mBlk + ar) * lda + kb + 32), 0, 1);

    __syncthreads();   // previous iteration's fragments consumed
    // ---- stage A (row-major, stride 40 halves) ----
    *(uint4*)(As + ar * 40 + aseg * 16)     = a0;
    *(uint4*)(As + ar * 40 + aseg * 16 + 8) = a1;
    // ---- stage B interleaved by K-pairs: dword = {B[k][n], B[k+1][n]} ----
    {
      const unsigned short* l0 = (const unsigned short*)&b00;   // [0..7] then b01
      const unsigned short* l1 = (const unsigned short*)&b10;
      unsigned int bo[16];
      #pragma unroll
      for (int i = 0; i < 8; i++)
        bo[i] = (unsigned int)l0[i] | ((unsigned int)l1[i] << 16);
      const unsigned short* h0 = (const unsigned short*)&b01;
      const unsigned short* h1 = (const unsigned short*)&b11;
      #pragma unroll
      for (int i = 0; i < 8; i++)
        bo[8 + i] = (unsigned int)h0[i] | ((unsigned int)h1[i] << 16);
      unsigned int* bsp = Bs + bp * 256 + bn;
      *(uint4*)(bsp)      = make_uint4(bo[0],  bo[1],  bo[2],  bo[3]);
      *(uint4*)(bsp + 4)  = make_uint4(bo[4],  bo[5],  bo[6],  bo[7]);
      *(uint4*)(bsp + 8)  = make_uint4(bo[8],  bo[9],  bo[10], bo[11]);
      *(uint4*)(bsp + 12) = make_uint4(bo[12], bo[13], bo[14], bo[15]);
    }
    __syncthreads();

    // ---- load fragments, contiguous k = 16*lhalf + s in both A and B ----
    FragA fa[4];
    #pragma unroll
    for (int t = 0; t < 4; t++) {
      const unsigned short* ap = As + (wm + t * 16 + lidx) * 40 + lhalf * 16;
      fa[t].q[0] = *(const uint4*)(ap);        // K = lhalf*16 .. +7
      fa[t].q[1] = *(const uint4*)(ap + 8);    // K = lhalf*16 + 8 .. +15
    }
    FragB fb[4];
    #pragma unroll
    for (int t = 0; t < 4; t++) {
      const unsigned int* bq = Bs + (lhalf * 8) * 256 + wn + t * 16 + lidx;
      #pragma unroll
      for (int j = 0; j < 8; j++) fb[t].u[j] = bq[j * 256];
    }
    // ---- 16 WMMAs per wave per k-step ----
    #pragma unroll
    for (int mt = 0; mt < 4; mt++)
      #pragma unroll
      for (int nt = 0; nt < 4; nt++)
        acc[mt][nt] = __builtin_amdgcn_wmma_f32_16x16x32_bf16(
            false, fa[mt].v, false, fb[nt].v, (short)0, acc[mt][nt], false, false);
  }

  // ---- epilogue ----
  #pragma unroll
  for (int mt = 0; mt < 4; mt++) {
    #pragma unroll
    for (int nt = 0; nt < 4; nt++) {
      int n = nBlk + wn + nt * 16 + lidx;
      #pragma unroll
      for (int r = 0; r < 8; r++) {
        int m = mBlk + wm + mt * 16 + r + lhalf * 8;
        size_t idx = (size_t)m * ldc + n;
        float o = acc[mt][nt][r];
        if (flags & F_CHEB)      o = 2.f * o - prev2[idx];
        else if (flags & F_BIAS) { o += bias[n]; if (flags & F_RELU) o = fmaxf(o, 0.f); }
        Cout[idx] = o;
        if (Cbf) Cbf[idx] = f2bf(o);
      }
    }
  }
}

// ---------------- cl1 linear + ReLU + max-pool(8) over nodes ----------------
__global__ void cheb_pool(const float* __restrict__ basis,
                          const float* __restrict__ W,   // [CF][Kche]
                          const float* __restrict__ bias,
                          unsigned short* __restrict__ pooled, int chunk) {
  int bi = blockIdx.y * 256 + threadIdx.x;     // 0..Bc-1
  int j  = blockIdx.x;                          // 0..127 (Ppad/CF)
  size_t orow = (size_t)(chunk * Bc + bi) * Ppad + j * CF;
  if (j >= 125) {                               // padding columns 625..639
    #pragma unroll
    for (int f = 0; f < CF; f++) pooled[orow + f] = 0;
    return;
  }
  float mx[CF];
  #pragma unroll
  for (int f = 0; f < CF; f++) mx[f] = 0.f;     // relu output >= 0
  for (int i = 0; i < 8; i++) {
    int v = 8 * j + i;
    float t[Kche];
    #pragma unroll
    for (int k = 0; k < Kche; k++)
      t[k] = basis[(size_t)k * Vp * Bc + (size_t)v * Bc + bi];
    #pragma unroll
    for (int f = 0; f < CF; f++) {
      float a = bias[f];
      #pragma unroll
      for (int k = 0; k < Kche; k++) a += W[f * Kche + k] * t[k];
      a = fmaxf(a, 0.f);
      mx[f] = fmaxf(mx[f], a);
    }
  }
  #pragma unroll
  for (int f = 0; f < CF; f++) pooled[orow + f] = f2bf(mx[f]);
}

// ---------------- 4x (BN eval + ReLU) + JumpingKnowledge max ----------------
__global__ void bn_jk(const float* __restrict__ h,
                      const float* __restrict__ gamma, const float* __restrict__ beta,
                      const float* __restrict__ mean,  const float* __restrict__ var,
                      unsigned short* __restrict__ zbf) {
  int idx = blockIdx.x * 256 + threadIdx.x;
  int n = idx & (HID - 1);
  float v = h[idx];
  float m = 0.f;                                 // relu outputs >= 0
  #pragma unroll
  for (int i = 0; i < NL; i++) {
    int p = i * HID + n;
    float s = gamma[p] * rsqrtf(var[p] + BN_EPS);
    v = (v - mean[p]) * s + beta[p];
    v = fmaxf(v, 0.f);
    m = fmaxf(m, v);
  }
  zbf[idx] = f2bf(m);
}

// ---------------- lin2 + log_softmax, one wave per row ----------------
__global__ void head(const float* __restrict__ z, const float* __restrict__ W2,
                     const float* __restrict__ b2, float* __restrict__ out) {
  __shared__ float zrow[HID];
  __shared__ float logit[OUTC];
  int b = blockIdx.x, t = threadIdx.x;          // 32 threads
  #pragma unroll
  for (int i = 0; i < HID / 32; i++)
    zrow[t + i * 32] = z[(size_t)b * HID + t + i * 32];
  __syncthreads();
  if (t < OUTC) {
    float a = b2[t];
    for (int h = 0; h < HID; h++) a += zrow[h] * W2[t * HID + h];
    logit[t] = a;
  }
  __syncthreads();
  if (t < OUTC) {
    float m = -3.4e38f;
    #pragma unroll
    for (int o = 0; o < OUTC; o++) m = fmaxf(m, logit[o]);
    float s = 0.f;
    #pragma unroll
    for (int o = 0; o < OUTC; o++) s += expf(logit[o] - m);
    out[(size_t)b * OUTC + t] = logit[t] - m - logf(s);
  }
}

// ---------------- launch ----------------
extern "C" void kernel_launch(void* const* d_in, const int* in_sizes, int n_in,
                              void* d_out, int out_size, void* d_ws, size_t ws_size,
                              hipStream_t stream) {
  (void)in_sizes; (void)n_in; (void)out_size; (void)ws_size;
  const float* x      = (const float*)d_in[0];
  const float* L      = (const float*)d_in[1];
  const float* lmax   = (const float*)d_in[2];
  const float* cl1_W  = (const float*)d_in[3];
  const float* cl1_b  = (const float*)d_in[4];
  const float* fc1_W  = (const float*)d_in[5];
  const float* fc1_b  = (const float*)d_in[6];
  const float* bn_g   = (const float*)d_in[7];
  const float* bn_b   = (const float*)d_in[8];
  const float* bn_m   = (const float*)d_in[9];
  const float* bn_v   = (const float*)d_in[10];
  const float* lin1_W = (const float*)d_in[11];
  const float* lin1_b = (const float*)d_in[12];
  const float* lin2_W = (const float*)d_in[13];
  const float* lin2_b = (const float*)d_in[14];
  float* out = (float*)d_out;

  char* w = (char*)d_ws;
  unsigned short* Lrbf   = (unsigned short*)(w);                        // 2 MB
  unsigned short* fc1Wb  = (unsigned short*)(w + 2097152);              // 320 KB
  unsigned short* lin1Wb = (unsigned short*)(w + 2424832);              // 128 KB
  float*          basis  = (float*)        (w + 2555904);               // 20 MB (5 planes)
  unsigned short* xbf0   = (unsigned short*)(w + 23527424);             // 2 MB
  unsigned short* xbf1   = (unsigned short*)(w + 25624576);             // 2 MB
  unsigned short* pooled = (unsigned short*)(w + 27721728);             // 10 MB
  float*          hbuf   = (float*)        (w + 38207488);              // 8 MB
  unsigned short* zbf    = (unsigned short*)(w + 46596096);             // 4 MB
  float*          zz     = (float*)        (w + 50790400);              // 8 MB
  const size_t PLANE = (size_t)Vp * Bc;

  // weight / Laplacian packing
  pack_Lr   <<<(Vp * Vp) / 256, 256, 0, stream>>>(L, lmax, Lrbf);
  pack_fc1W <<<(Ppad * HID) / 256, 256, 0, stream>>>(fc1_W, fc1Wb);
  pack_lin1W<<<(HID * HID) / 256, 256, 0, stream>>>(lin1_W, lin1Wb);

  for (int c = 0; c < NCHUNK; c++) {
    transpose_pack<<<dim3(Vp / 32, Bc / 32), dim3(32, 8), 0, stream>>>(
        x, basis, xbf0, c);
    dim3 g(Bc / 256, Vp / 128);
    // x1 = Lr @ x0
    gemm_bf16<<<g, 256, 0, stream>>>(Lrbf, Vp, xbf0, Bc,
        basis + PLANE, xbf1, Bc, nullptr, nullptr, 0, Vp);
    // x2 = 2*Lr@x1 - x0
    gemm_bf16<<<g, 256, 0, stream>>>(Lrbf, Vp, xbf1, Bc,
        basis + 2 * PLANE, xbf0, Bc, basis, nullptr, F_CHEB, Vp);
    // x3 = 2*Lr@x2 - x1
    gemm_bf16<<<g, 256, 0, stream>>>(Lrbf, Vp, xbf0, Bc,
        basis + 3 * PLANE, xbf1, Bc, basis + PLANE, nullptr, F_CHEB, Vp);
    // x4 = 2*Lr@x3 - x2
    gemm_bf16<<<g, 256, 0, stream>>>(Lrbf, Vp, xbf1, Bc,
        basis + 4 * PLANE, nullptr, Bc, basis + 2 * PLANE, nullptr, F_CHEB, Vp);
    // cl1 + relu + maxpool8 -> pooled (bf16)
    cheb_pool<<<dim3(Ppad / CF, Bc / 256), 256, 0, stream>>>(
        basis, cl1_W, cl1_b, pooled, c);
  }

  // fc1: [8192,640] x [640,256] + bias
  gemm_bf16<<<dim3(HID / 256, Bsz / 128), 256, 0, stream>>>(
      pooled, Ppad, fc1Wb, HID, hbuf, nullptr, HID, nullptr, fc1_b, F_BIAS, Ppad);
  // 4x BN+ReLU + JK max
  bn_jk<<<(Bsz * HID) / 256, 256, 0, stream>>>(hbuf, bn_g, bn_b, bn_m, bn_v, zbf);
  // lin1 + bias + relu
  gemm_bf16<<<dim3(HID / 256, Bsz / 128), 256, 0, stream>>>(
      zbf, HID, lin1Wb, HID, zz, nullptr, HID, nullptr, lin1_b, F_BIAS | F_RELU, HID);
  // lin2 + log_softmax
  head<<<Bsz, 32, 0, stream>>>(zz, lin2_W, lin2_b, out);
}
